// VQ_17437567222444
// MI455X (gfx1250) — matrix operand order repaired and checked by reading, not compiled, not run on hardware
//
#include <hip/hip_runtime.h>

#define CDIM   256
#define KCODES 4096
#define NROWS  32768      // B*W*H = 32*32*32
#define MBLK   128        // rows per workgroup
#define NCH    128        // codes per chunk
#define XPITCH 264        // 256 + 8 pad (bf16 elems) -> conflict-free b128 reads
#define BPITCH 136        // 128 + 8 pad (bf16 elems) -> conflict-free b128 reads

// LDS partition byte offsets (dynamic-LDS only => base offset 0)
#define OFF_XH  0
#define OFF_XL  (MBLK * XPITCH * 2)
#define OFF_BH  (2 * MBLK * XPITCH * 2)                 // 135168
#define OFF_BL  (OFF_BH + CDIM * BPITCH * 2)            // +69632
#define OFF_RED (OFF_BL + CDIM * BPITCH * 2)            // +69632
#define SMEM_BYTES (OFF_RED + 2 * MBLK * 8)             // 276480

typedef __attribute__((ext_vector_type(16))) __bf16 bf16x16;
typedef __attribute__((ext_vector_type(8)))  float  f32x8;

union Frag { uint4 q[2]; bf16x16 v; };

__device__ __forceinline__ unsigned short f2bf(float f) {
  unsigned int u = __float_as_uint(f);
  u += 0x7FFFu + ((u >> 16) & 1u);          // round-to-nearest-even
  return (unsigned short)(u >> 16);
}
__device__ __forceinline__ float bf2f(unsigned short h) {
  return __uint_as_float(((unsigned int)h) << 16);
}

// Issue one chunk's codebook tile as async DMA global->LDS (32 x b128 / thread).
// Thread t copies channel-row t of both hi and lo arrays (256B each).
__device__ __forceinline__ void issue_b_chunk(const unsigned short* ch,
                                              const unsigned short* cl,
                                              int cbase, int tid) {
  unsigned ldsA = OFF_BH + (unsigned)tid * (BPITCH * 2);
  unsigned ldsB = OFF_BL + (unsigned)tid * (BPITCH * 2);
  unsigned gofs = (unsigned)((tid * KCODES + cbase) * 2);
  #pragma unroll
  for (int kk = 0; kk < 16; ++kk) {
    asm volatile("global_load_async_to_lds_b128 %0, %1, %2"
                 :: "v"(ldsA), "v"(gofs), "s"(ch) : "memory");
    asm volatile("global_load_async_to_lds_b128 %0, %1, %2"
                 :: "v"(ldsB), "v"(gofs), "s"(cl) : "memory");
    ldsA += 16; ldsB += 16; gofs += 16;
  }
}

// ---- Kernel 1a: x [B,C,H,W] fp32 -> xt_hi/lo [n][c] bf16 (n = (b*W+w)*H+h) ----
__global__ void vq_split_x(const float* __restrict__ x,
                           unsigned short* __restrict__ xh,
                           unsigned short* __restrict__ xl) {
  int g = blockIdx.x * 256 + threadIdx.x;   // 0 .. NROWS*CDIM-1
  int n = g >> 8, c = g & 255;
  int b = n >> 10, w = (n >> 5) & 31, h = n & 31;
  float v = x[((b * 256 + c) * 32 + h) * 32 + w];
  unsigned short hi = f2bf(v);
  xh[g] = hi;
  xl[g] = f2bf(v - bf2f(hi));
}

// ---- Kernel 1b: codebook [K,C] -> transposed cbt_hi/lo [c][K] + ||e||^2 ----
__global__ void vq_split_cb(const float* __restrict__ cb,
                            unsigned short* __restrict__ ch,
                            unsigned short* __restrict__ cl,
                            float* __restrict__ enorm) {
  int k = blockIdx.x, c = threadIdx.x;
  float v = cb[k * CDIM + c];
  unsigned short hi = f2bf(v);
  ch[c * KCODES + k] = hi;
  cl[c * KCODES + k] = f2bf(v - bf2f(hi));
  __shared__ float sm[256];
  sm[c] = v * v;
  __syncthreads();
  for (int s = 128; s > 0; s >>= 1) {
    if (c < s) sm[c] += sm[c + s];
    __syncthreads();
  }
  if (c == 0) enorm[k] = sm[0];
}

// ---- Kernel 2: fused bf16x3 WMMA GEMM + running argmin over codes ----
__global__ void __launch_bounds__(256)
vq_argmin_kernel(const unsigned short* __restrict__ xh,
                 const unsigned short* __restrict__ xl,
                 const unsigned short* __restrict__ ch,
                 const unsigned short* __restrict__ cl,
                 const float* __restrict__ enorm,
                 int* __restrict__ idx_out) {
  extern __shared__ __align__(16) unsigned short smem[];
  unsigned short* Xh = smem;                            // 128 x 264 bf16
  unsigned short* Xl = (unsigned short*)((char*)smem + OFF_XL);
  unsigned short* Bh = (unsigned short*)((char*)smem + OFF_BH);  // 256 x 136 bf16
  unsigned short* Bl = (unsigned short*)((char*)smem + OFF_BL);
  float* redD = (float*)((char*)smem + OFF_RED);        // 2 x 128
  int*   redC = (int*)(redD + 2 * MBLK);                // 2 x 128

  const int tid  = threadIdx.x;
  const int lane = tid & 31;
  const int wid  = tid >> 5;
  const int waveM = wid >> 1;                   // 4 M-waves: 32-row groups
  const int waveN = wid & 1;                    // 2 N-waves: 64-code groups
  const int nbase = blockIdx.x * MBLK;

  // stage X tile once (sync): contiguous 16B copies
  for (int s = tid; s < MBLK * 32; s += 256) {
    int row = s >> 5, part = s & 31;
    *(uint4*)(Xh + row * XPITCH + part * 8) =
        *(const uint4*)(xh + (nbase + row) * CDIM + part * 8);
    *(uint4*)(Xl + row * XPITCH + part * 8) =
        *(const uint4*)(xl + (nbase + row) * CDIM + part * 8);
  }

  // kick off async DMA of the first codebook chunk
  issue_b_chunk(ch, cl, 0, tid);

  float bestD[16];
  int   bestC[16];
  #pragma unroll
  for (int i = 0; i < 16; ++i) { bestD[i] = 3.4e38f; bestC[i] = 0; }

  const f32x8 vzero = {0.f,0.f,0.f,0.f,0.f,0.f,0.f,0.f};
  f32x8 acc[2][4];
  #pragma unroll
  for (int mt = 0; mt < 2; ++mt)
    #pragma unroll
    for (int nt = 0; nt < 4; ++nt) acc[mt][nt] = vzero;

  const int m     = lane & 15;                  // A: M index / argmin column
  const int kh8   = (lane >> 4) * 8;            // A: K-half select
  const int mrow0 = waveM * 32;
  const int ncol0 = waveN * 64;

  for (int ci = 0; ci < KCODES / NCH; ++ci) {
    const int cbase = ci * NCH;

    // chunk data becomes visible: own DMAs done, then all waves' DMAs done
    asm volatile("s_wait_asynccnt 0x0" ::: "memory");
    __syncthreads();

    float en[4];
    #pragma unroll
    for (int nt = 0; nt < 4; ++nt)
      en[nt] = enorm[cbase + ncol0 + nt * 16 + m];

    #pragma unroll
    for (int ks = 0; ks < 8; ++ks) {
      const int kb = ks * 32;
      Frag ah[2], al[2], bhf[4], blf[4];
      #pragma unroll
      for (int mt = 0; mt < 2; ++mt) {          // A 16x32 bf16 lane layout
        const unsigned short* ph = Xh + (mrow0 + mt * 16 + m) * XPITCH + kb + kh8;
        ah[mt].q[0] = *(const uint4*)ph;
        ah[mt].q[1] = *(const uint4*)(ph + 16);
        const unsigned short* pl = Xl + (mrow0 + mt * 16 + m) * XPITCH + kb + kh8;
        al[mt].q[0] = *(const uint4*)pl;
        al[mt].q[1] = *(const uint4*)(pl + 16);
      }
      #pragma unroll
      for (int nt = 0; nt < 4; ++nt) {          // B 32x16 bf16 lane layout
        const unsigned short* ph = Bh + (kb + kh8) * BPITCH + ncol0 + nt * 16 + m;
        const unsigned short* pl = Bl + (kb + kh8) * BPITCH + ncol0 + nt * 16 + m;
        #pragma unroll
        for (int s = 0; s < 16; ++s) {
          ((unsigned short*)bhf[nt].q)[s] = ph[((s & 7) + (s >> 3) * 16) * BPITCH];
          ((unsigned short*)blf[nt].q)[s] = pl[((s & 7) + (s >> 3) * 16) * BPITCH];
        }
      }
      #pragma unroll
      for (int mt = 0; mt < 2; ++mt)
        #pragma unroll
        for (int nt = 0; nt < 4; ++nt) {        // bf16x3: hh + hl + lh
          acc[mt][nt] = __builtin_amdgcn_wmma_f32_16x16x32_bf16(
              false, ah[mt].v, false, bhf[nt].v, (short)0, acc[mt][nt], false, false);
          acc[mt][nt] = __builtin_amdgcn_wmma_f32_16x16x32_bf16(
              false, ah[mt].v, false, blf[nt].v, (short)0, acc[mt][nt], false, false);
          acc[mt][nt] = __builtin_amdgcn_wmma_f32_16x16x32_bf16(
              false, al[mt].v, false, bhf[nt].v, (short)0, acc[mt][nt], false, false);
        }
    }

    // everyone finished reading B for this chunk -> refill buffer for the next
    __syncthreads();
    if (ci + 1 < KCODES / NCH) issue_b_chunk(ch, cl, cbase + NCH, tid);

    // register-only argmin fold: overlaps the in-flight DMA
    #pragma unroll
    for (int nt = 0; nt < 4; ++nt) {
      int code = cbase + ncol0 + nt * 16 + m;
      #pragma unroll
      for (int mt = 0; mt < 2; ++mt)
        #pragma unroll
        for (int r = 0; r < 8; ++r) {
          float d = fmaf(-2.0f, acc[mt][nt][r], en[nt]);
          int slot = mt * 8 + r;
          if (d < bestD[slot]) { bestD[slot] = d; bestC[slot] = code; }
          acc[mt][nt][r] = 0.0f;
        }
    }
  }

  // cross-lane min within each 16-lane half (same rows, different code columns)
  #pragma unroll
  for (int slot = 0; slot < 16; ++slot) {
    float d = bestD[slot]; int c = bestC[slot];
    #pragma unroll
    for (int off = 1; off < 16; off <<= 1) {
      float od = __shfl_xor(d, off, 32);
      int   oc = __shfl_xor(c, off, 32);
      if (od < d) { d = od; c = oc; }
    }
    bestD[slot] = d; bestC[slot] = c;
  }

  __syncthreads();
  if ((lane & 15) == 0) {
    int half8 = (lane >> 4) * 8;                // lane0: M=+0..7, lane16: M=+8..15
    #pragma unroll
    for (int mt = 0; mt < 2; ++mt)
      #pragma unroll
      for (int r = 0; r < 8; ++r) {
        int row = mrow0 + mt * 16 + half8 + r;
        redD[waveN * MBLK + row] = bestD[mt * 8 + r];
        redC[waveN * MBLK + row] = bestC[mt * 8 + r];
      }
  }
  __syncthreads();
  if (tid < MBLK) {
    float d0 = redD[tid], d1 = redD[MBLK + tid];
    idx_out[nbase + tid] = (d1 < d0) ? redC[MBLK + tid] : redC[tid];
  }
}

// ---- Kernel 3: gather codes back to [B,C,H,W] (codebook is L2-resident) ----
__global__ void vq_gather(const float* __restrict__ cb,
                          const int* __restrict__ idx,
                          float* __restrict__ out) {
  int g = blockIdx.x * 256 + threadIdx.x;       // linear over B*C*H*W
  int b = g >> 18, c = (g >> 10) & 255, h = (g >> 5) & 31, w = g & 31;
  int n = (b << 10) + (w << 5) + h;
  out[g] = cb[idx[n] * CDIM + c];
}

// ---- Kernel 4: indices [B,H,W] (as float, harness output convention) ----
__global__ void vq_indices(const int* __restrict__ idx,
                           float* __restrict__ out) {
  int t = blockIdx.x * 256 + threadIdx.x;       // linear over B*H*W
  int b = t >> 10, h = (t >> 5) & 31, w = t & 31;
  int n = (b << 10) + (w << 5) + h;
  out[t] = (float)idx[n];
}

extern "C" void kernel_launch(void* const* d_in, const int* in_sizes, int n_in,
                              void* d_out, int out_size, void* d_ws, size_t ws_size,
                              hipStream_t stream) {
  const float* x  = (const float*)d_in[0];      // [32,256,32,32] fp32
  const float* cb = (const float*)d_in[1];      // [4096,256]     fp32
  float* out_codes = (float*)d_out;             // 8388608 floats
  float* out_idx   = out_codes + (NROWS * CDIM);// + 32768 floats

  // workspace layout (~37.9 MB)
  unsigned short* xt_hi  = (unsigned short*)d_ws;
  unsigned short* xt_lo  = xt_hi  + (size_t)NROWS * CDIM;
  unsigned short* cbt_hi = xt_lo  + (size_t)NROWS * CDIM;
  unsigned short* cbt_lo = cbt_hi + (size_t)KCODES * CDIM;
  float*          enorm  = (float*)(cbt_lo + (size_t)KCODES * CDIM);
  int*            idx    = (int*)(enorm + KCODES);

  hipFuncSetAttribute((const void*)vq_argmin_kernel,
                      hipFuncAttributeMaxDynamicSharedMemorySize, SMEM_BYTES);

  vq_split_x <<<NROWS * CDIM / 256, 256, 0, stream>>>(x, xt_hi, xt_lo);
  vq_split_cb<<<KCODES, 256, 0, stream>>>(cb, cbt_hi, cbt_lo, enorm);
  vq_argmin_kernel<<<NROWS / MBLK, 256, SMEM_BYTES, stream>>>(xt_hi, xt_lo,
                                                              cbt_hi, cbt_lo,
                                                              enorm, idx);
  vq_gather  <<<NROWS * CDIM / 256, 256, 0, stream>>>(cb, idx, out_codes);
  vq_indices <<<NROWS / 256, 256, 0, stream>>>(idx, out_idx);
}
